// MultiHeadAttention_54683523612897
// MI455X (gfx1250) — compile-verified, matrix operand order checked by reading, not run on hardware
//
#include <hip/hip_runtime.h>
#include <hip/hip_bf16.h>

// ---------- problem constants ----------
#define BATCH     4
#define SEQ       1024
#define EMBED     1024
#define HEADS     16
#define HDIM      64           // EMBED / HEADS
#define ODIM      512
#define MTOT      (BATCH * SEQ)        // 4096 rows for all projection GEMMs

typedef __attribute__((ext_vector_type(16))) __bf16 v16bf;
typedef __attribute__((ext_vector_type(8)))  float  v8f;
typedef __attribute__((ext_vector_type(4)))  unsigned int v4u;
typedef __attribute__((ext_vector_type(8)))  int  v8i;
typedef __attribute__((ext_vector_type(4)))  int  v4i;

union bfrag { v16bf v; uint4 u[2]; };

#if __has_builtin(__builtin_amdgcn_tensor_load_to_lds) && \
    __has_builtin(__builtin_amdgcn_s_wait_tensorcnt)
#define HAVE_TDM 1
#else
#define HAVE_TDM 0
#endif

// Load a 16x32 bf16 WMMA A/B fragment (K-major source, ld = row stride in elems).
// ISA layout: lanes 0-15 hold row (row0+lane), K chunks {k..k+7, k+16..k+23};
// lanes 16-31 hold the same rows, K chunks {k+8..k+15, k+24..k+31}.
__device__ __forceinline__ v16bf load_frag(const __bf16* __restrict__ base,
                                           int ld, int row0, int k, int lane) {
  const __bf16* p = base + (size_t)(row0 + (lane & 15)) * ld + (k + ((lane >> 4) << 3));
  bfrag f;
  f.u[0] = *(const uint4*)(p);
  f.u[1] = *(const uint4*)(p + 16);
  return f.v;
}

__device__ __forceinline__ v8f wmma_bf16(v16bf a, v16bf b, v8f c) {
  return __builtin_amdgcn_wmma_f32_16x16x32_bf16(false, a, false, b, (short)0, c,
                                                 false, false);
}

#if HAVE_TDM
// TDM: DMA a 2D bf16 tile [tile_rows x tile_k] (row stride = row_stride elems)
// from global memory into LDS at lds_byte_addr (packed [tile_rows][tile_k]).
// D# layout per CDNA5 ISA.
__device__ __forceinline__ void tdm_load_2d_bf16(const void* gaddr,
                                                 unsigned lds_byte_addr,
                                                 unsigned tile_k, unsigned tile_rows,
                                                 unsigned tensor_d0, unsigned tensor_d1,
                                                 unsigned row_stride) {
  unsigned long long ga = (unsigned long long)gaddr;
  v4u g0;
  g0[0] = 1u;                                            // count=1, user mode
  g0[1] = lds_byte_addr;                                 // lds_addr
  g0[2] = (unsigned)(ga & 0xffffffffu);                  // global_addr[31:0]
  g0[3] = (unsigned)((ga >> 32) & 0x1ffffffu) | (2u << 30); // addr[56:32] | type=2
  v8i g1;
  g1[0] = (int)(1u << 16);                               // data_size=1 (2 bytes)
  g1[1] = (int)((tensor_d0 & 0xffffu) << 16);            // tensor_dim0[15:0]
  g1[2] = (int)(((tensor_d0 >> 16) & 0xffffu) |
                ((tensor_d1 & 0xffffu) << 16));          // dim0[31:16] | dim1[15:0]
  g1[3] = (int)(((tensor_d1 >> 16) & 0xffffu) |
                ((tile_k & 0xffffu) << 16));             // dim1[31:16] | tile_dim0
  g1[4] = (int)(tile_rows & 0xffffu);                    // tile_dim1 (tile_dim2=0)
  g1[5] = (int)row_stride;                               // dim0_stride[31:0]
  g1[6] = 0;                                             // dim0_stride[47:32]
  g1[7] = 0;
  v4i z4 = {0, 0, 0, 0};
#if __clang_major__ >= 23
  v8i z8 = {0, 0, 0, 0, 0, 0, 0, 0};
  __builtin_amdgcn_tensor_load_to_lds(g0, g1, z4, z4, z8, 0);
#else
  __builtin_amdgcn_tensor_load_to_lds(g0, g1, z4, z4, 0);
#endif
}
#endif

// ---------- f32 -> bf16 conversion (8 elems / thread, b128 traffic) ----------
__global__ __launch_bounds__(256) void cvt_f32_bf16(const float* __restrict__ in,
                                                    __bf16* __restrict__ out, int n) {
  const int i = (blockIdx.x * 256 + threadIdx.x) * 8;
  if (i >= n) return;
  const float4 f0 = *(const float4*)(in + i);
  const float4 f1 = *(const float4*)(in + i + 4);
  union { __bf16 h[8]; uint4 u; } o;
  o.h[0] = (__bf16)f0.x; o.h[1] = (__bf16)f0.y; o.h[2] = (__bf16)f0.z; o.h[3] = (__bf16)f0.w;
  o.h[4] = (__bf16)f1.x; o.h[5] = (__bf16)f1.y; o.h[6] = (__bf16)f1.z; o.h[7] = (__bf16)f1.w;
  *(uint4*)(out + i) = o.u;
}

// ---------- generic NT GEMM: C[M,N] = A[M,K] * B[N,K]^T + bias ----------
// Block: 256 threads = 8 wave32s; tile 256(M) x 64(N); each wave owns 32 rows.
// B panel (64 x 32 bf16) staged in LDS via TDM (double-buffered, TENSORcnt).
// MODE 0: f32 row-major out  | MODE 1: bf16 row-major out
// MODE 2: bf16 -> Q/K [B,H,S,D] | MODE 3: bf16 -> Vt [B,H,D,S]
template <int MODE>
__global__ __launch_bounds__(256)
void gemm_bf16_nt(const __bf16* __restrict__ A, const __bf16* __restrict__ Bw,
                  const float* __restrict__ bias, float* __restrict__ Cf,
                  __bf16* __restrict__ Cb, int M, int N, int K) {
  __shared__ __bf16 Bs[2][64 * 32];            // two 4 KB B panels
  const int tid  = threadIdx.x;
  const int lane = tid & 31;
  const int wave = tid >> 5;
  const int m0   = blockIdx.x * 256 + wave * 32;
  const int n0   = blockIdx.y * 64;

  v8f acc[8] = {};                             // [2 m-subtiles][4 n-tiles]

#if HAVE_TDM
  if (wave == 0)
    tdm_load_2d_bf16(Bw + (size_t)n0 * K, (unsigned)(unsigned long long)&Bs[0][0],
                     32, 64, (unsigned)K, (unsigned)N, (unsigned)K);
#endif

  for (int k = 0; k < K; k += 32) {
    const int cur = (k >> 5) & 1;
#if HAVE_TDM
    if (wave == 0) {
      if (k + 32 < K) {
        tdm_load_2d_bf16(Bw + (size_t)n0 * K + (k + 32),
                         (unsigned)(unsigned long long)&Bs[cur ^ 1][0],
                         32, 64, (unsigned)K, (unsigned)N, (unsigned)K);
        __builtin_amdgcn_s_wait_tensorcnt(1);  // current panel done, next in flight
      } else {
        __builtin_amdgcn_s_wait_tensorcnt(0);  // drain for last panel
      }
    }
    __syncthreads();
#else
    {                                          // cooperative fallback staging
      const int r = tid >> 2, c = (tid & 3) << 3;
      *(uint4*)(&Bs[cur][r * 32 + c]) =
          *(const uint4*)(Bw + (size_t)(n0 + r) * K + k + c);
    }
    __syncthreads();
#endif

    if (k + 32 < K)
      __builtin_prefetch(A + (size_t)(m0 + (lane & 15)) * K + k + 32, 0, 1);

    const v16bf a0 = load_frag(A, K, m0,      k, lane);
    const v16bf a1 = load_frag(A, K, m0 + 16, k, lane);
#pragma unroll
    for (int t = 0; t < 4; ++t) {
      const __bf16* bp = &Bs[cur][((t << 4) + (lane & 15)) * 32 + ((lane >> 4) << 3)];
      bfrag bf;
      bf.u[0] = *(const uint4*)(bp);
      bf.u[1] = *(const uint4*)(bp + 16);
      acc[t]     = wmma_bf16(a0, bf.v, acc[t]);
      acc[4 + t] = wmma_bf16(a1, bf.v, acc[4 + t]);
    }
    __syncthreads();
  }

  // epilogue: C layout -> lane holds column n, VGPR i holds row (msub, mbase+i)
  const int nlane = lane & 15;
  const int mbase = (lane >> 4) << 3;
#pragma unroll
  for (int sm = 0; sm < 2; ++sm) {
#pragma unroll
    for (int t = 0; t < 4; ++t) {
      const int n  = n0 + (t << 4) + nlane;
      const float bv = bias[n];
#pragma unroll
      for (int i = 0; i < 8; ++i) {
        const int m = m0 + sm * 16 + mbase + i;
        const float val = acc[sm * 4 + t][i] + bv;
        if (MODE == 0) {
          Cf[(size_t)m * N + n] = val;
        } else if (MODE == 1) {
          Cb[(size_t)m * N + n] = (__bf16)val;
        } else if (MODE == 2) {               // [B,H,S,D]
          const int b = m >> 10, s = m & (SEQ - 1);
          const int h = n >> 6,  d = n & (HDIM - 1);
          Cb[(((size_t)b * HEADS + h) * SEQ + s) * HDIM + d] = (__bf16)val;
        } else {                              // MODE 3: [B,H,D,S]
          const int b = m >> 10, s = m & (SEQ - 1);
          const int h = n >> 6,  d = n & (HDIM - 1);
          Cb[(((size_t)b * HEADS + h) * HDIM + d) * SEQ + s] = (__bf16)val;
        }
      }
    }
  }
}

// ---------- scores = softmax(Q K^T / sqrt(D)) ----------
// grid: (SEQ/32, BATCH*HEADS); 256 threads; 128 KB dynamic LDS (32 x 1024 f32)
__global__ __launch_bounds__(256)
void attn_scores_softmax(const __bf16* __restrict__ Qb, const __bf16* __restrict__ Kb,
                         float* __restrict__ attn) {
  extern __shared__ float sc[];               // [32][SEQ]
  const int bh  = blockIdx.y;
  const int q0  = blockIdx.x * 32;
  const int tid = threadIdx.x, lane = tid & 31, wave = tid >> 5;
  const int qb  = wave >> 2;                  // which 16-row half
  const int cs  = wave & 3;                   // 256-column strip

  const __bf16* Qh = Qb + (size_t)bh * SEQ * HDIM;
  const __bf16* Kh = Kb + (size_t)bh * SEQ * HDIM;

  const v16bf a0 = load_frag(Qh, HDIM, q0 + qb * 16, 0, lane);
  const v16bf a1 = load_frag(Qh, HDIM, q0 + qb * 16, 32, lane);

  const int nlane = lane & 15;
  const int mbase = (lane >> 4) << 3;

  // software pipeline: prefetch next K tile while computing current
  v16bf b0 = load_frag(Kh, HDIM, cs * 256, 0, lane);
  v16bf b1 = load_frag(Kh, HDIM, cs * 256, 32, lane);
#pragma unroll 4
  for (int t = 0; t < 16; ++t) {
    const int n0 = cs * 256 + (t << 4);
    v16bf nb0 = b0, nb1 = b1;
    if (t < 15) {
      nb0 = load_frag(Kh, HDIM, n0 + 16, 0, lane);
      nb1 = load_frag(Kh, HDIM, n0 + 16, 32, lane);
    }
    v8f c = {};
    c = wmma_bf16(a0, b0, c);
    c = wmma_bf16(a1, b1, c);
    const int n = n0 + nlane;
#pragma unroll
    for (int i = 0; i < 8; ++i)
      sc[(qb * 16 + mbase + i) * SEQ + n] = c[i] * 0.125f;   // 1/sqrt(64)
    b0 = nb0; b1 = nb1;
  }
  __syncthreads();

  // softmax: 8 threads per row, 128 cols each, wave32 shfl reductions
  const int row = tid >> 3, seg = tid & 7;
  float* r = sc + row * SEQ + seg * 128;
  float mx = -3.4e38f;
#pragma unroll 8
  for (int j = 0; j < 128; ++j) mx = fmaxf(mx, r[j]);
#pragma unroll
  for (int o = 1; o < 8; o <<= 1) mx = fmaxf(mx, __shfl_xor(mx, o, 32));
  float sum = 0.f;
#pragma unroll 8
  for (int j = 0; j < 128; ++j) { float e = __expf(r[j] - mx); r[j] = e; sum += e; }
#pragma unroll
  for (int o = 1; o < 8; o <<= 1) sum += __shfl_xor(sum, o, 32);
  const float rinv = 1.f / sum;
  float* dst = attn + ((size_t)bh * SEQ + (q0 + row)) * SEQ + seg * 128;
#pragma unroll 8
  for (int j = 0; j < 128; ++j) dst[j] = r[j] * rinv;
}

// ---------- out = attn @ V  (written merged-head [B*S, E] bf16) ----------
// grid: (SEQ/128, BATCH*HEADS); 256 threads
__global__ __launch_bounds__(256)
void attn_v(const float* __restrict__ attn, const __bf16* __restrict__ Vt,
            __bf16* __restrict__ outb) {
  const int bh = blockIdx.y, b = bh >> 4, h = bh & 15;
  const int tid = threadIdx.x, lane = tid & 31, wave = tid >> 5;
  const int m0 = blockIdx.x * 128 + wave * 16;   // q-row within this head

  const float*  Ah = attn + (size_t)bh * SEQ * SEQ;
  const __bf16* Vh = Vt   + (size_t)bh * HDIM * SEQ;

  v8f acc[4] = {};                               // D = 64 -> 4 n-tiles

  for (int k = 0; k < SEQ; k += 32) {
    // A fragment: f32 attn -> bf16 on the fly
    const float* p = Ah + (size_t)(m0 + (lane & 15)) * SEQ + k + ((lane >> 4) << 3);
    const float4 f0 = *(const float4*)(p);
    const float4 f1 = *(const float4*)(p + 4);
    const float4 f2 = *(const float4*)(p + 16);
    const float4 f3 = *(const float4*)(p + 20);
    const float tmp[16] = {f0.x, f0.y, f0.z, f0.w, f1.x, f1.y, f1.z, f1.w,
                           f2.x, f2.y, f2.z, f2.w, f3.x, f3.y, f3.z, f3.w};
    v16bf a;
#pragma unroll
    for (int i = 0; i < 16; ++i) a[i] = (__bf16)tmp[i];

    if (k + 32 < SEQ) __builtin_prefetch(p + 32, 0, 1);

#pragma unroll
    for (int t = 0; t < 4; ++t) {
      const v16bf bv = load_frag(Vh, SEQ, t << 4, k, lane);
      acc[t] = wmma_bf16(a, bv, acc[t]);
    }
  }

  const int nlane = lane & 15;
  const int mbase = (lane >> 4) << 3;
#pragma unroll
  for (int t = 0; t < 4; ++t) {
    const int e = h * HDIM + (t << 4) + nlane;
#pragma unroll
    for (int i = 0; i < 8; ++i) {
      const int s = m0 + mbase + i;
      outb[((size_t)b * SEQ + s) * EMBED + e] = (__bf16)acc[t][i];
    }
  }
}

// ---------- host-side orchestration ----------
static inline char* align_up(char* p, size_t a) {
  return (char*)(((uintptr_t)p + a - 1) & ~(uintptr_t)(a - 1));
}

extern "C" void kernel_launch(void* const* d_in, const int* in_sizes, int n_in,
                              void* d_out, int out_size, void* d_ws, size_t ws_size,
                              hipStream_t stream) {
  const float* x  = (const float*)d_in[0];
  // d_in[1] = pre_data, size 0 -> no blend (static in reference)
  const float* Wq = (const float*)d_in[2];
  const float* bq = (const float*)d_in[3];
  const float* Wk = (const float*)d_in[4];
  const float* bk = (const float*)d_in[5];
  const float* Wv = (const float*)d_in[6];
  const float* bv = (const float*)d_in[7];
  const float* Wo = (const float*)d_in[8];
  const float* bo = (const float*)d_in[9];
  const float* Wd = (const float*)d_in[10];
  const float* bd = (const float*)d_in[11];

  float* out_f  = (float*)d_out;                       // [4096, 512]
  float* attn_f = (float*)d_out + (size_t)MTOT * ODIM; // [B,H,S,S]

  // carve workspace (all bf16, 256B aligned)
  char* w = (char*)d_ws;
  auto carve = [&](size_t elems) {
    w = align_up(w, 256);
    __bf16* p = (__bf16*)w;
    w += elems * sizeof(__bf16);
    return p;
  };
  __bf16* xb   = carve((size_t)MTOT * EMBED);
  __bf16* wq_b = carve((size_t)EMBED * EMBED);
  __bf16* wk_b = carve((size_t)EMBED * EMBED);
  __bf16* wv_b = carve((size_t)EMBED * EMBED);
  __bf16* wo_b = carve((size_t)EMBED * EMBED);
  __bf16* wd_b = carve((size_t)ODIM * EMBED);
  __bf16* Qb   = carve((size_t)MTOT * EMBED);   // [B,H,S,D]
  __bf16* Kb   = carve((size_t)MTOT * EMBED);   // [B,H,S,D]
  __bf16* Vtb  = carve((size_t)MTOT * EMBED);   // [B,H,D,S]
  __bf16* avob = carve((size_t)MTOT * EMBED);   // attn@V merged heads [B*S, E]
  __bf16* opjb = carve((size_t)MTOT * EMBED);   // out-projection result
  (void)ws_size;

  // 1) precision conversion (8 elems / thread)
  auto cvt = [&](const float* src, __bf16* dst, int n) {
    cvt_f32_bf16<<<(n / 8 + 255) / 256, 256, 0, stream>>>(src, dst, n);
  };
  cvt(x,  xb,   MTOT * EMBED);
  cvt(Wq, wq_b, EMBED * EMBED);
  cvt(Wk, wk_b, EMBED * EMBED);
  cvt(Wv, wv_b, EMBED * EMBED);
  cvt(Wo, wo_b, EMBED * EMBED);
  cvt(Wd, wd_b, ODIM * EMBED);

  // 2) QKV projections (head-split epilogues; V stored transposed per head)
  dim3 gproj(MTOT / 256, EMBED / 64);
  gemm_bf16_nt<2><<<gproj, 256, 0, stream>>>(xb, wq_b, bq, nullptr, Qb,
                                             MTOT, EMBED, EMBED);
  gemm_bf16_nt<2><<<gproj, 256, 0, stream>>>(xb, wk_b, bk, nullptr, Kb,
                                             MTOT, EMBED, EMBED);
  gemm_bf16_nt<3><<<gproj, 256, 0, stream>>>(xb, wv_b, bv, nullptr, Vtb,
                                             MTOT, EMBED, EMBED);

  // 3) scores + softmax -> attn (second output region of d_out)
  attn_scores_softmax<<<dim3(SEQ / 32, BATCH * HEADS), 256,
                        32 * SEQ * sizeof(float), stream>>>(Qb, Kb, attn_f);

  // 4) attn @ V -> merged-head activation (bf16)
  attn_v<<<dim3(SEQ / 128, BATCH * HEADS), 256, 0, stream>>>(attn_f, Vtb, avob);

  // 5) output projection (bf16 intermediate)
  gemm_bf16_nt<1><<<gproj, 256, 0, stream>>>(avob, wo_b, bo, nullptr, opjb,
                                             MTOT, EMBED, EMBED);

  // 6) final dense -> f32 out (first output region of d_out)
  gemm_bf16_nt<0><<<dim3(MTOT / 256, ODIM / 64), 256, 0, stream>>>(
      opjb, wd_b, bd, out_f, nullptr, MTOT, ODIM, EMBED);
}
